// TradeWealthModel_25829933318706
// MI455X (gfx1250) — compile-verified, matrix operand order checked by reading, not run on hardware
//
#include <hip/hip_runtime.h>

typedef __attribute__((ext_vector_type(16))) _Float16 v16h;
typedef __attribute__((ext_vector_type(8)))  float    v8f;

#define NSTEPS 20
#define MDIM   64

// One wave (32 lanes) owns 16 rows of K. Lane l: row r = l&15, half g = l>>4.
// State layout per lane (matches WMMA B-frag AND D-frag layouts):
//   pr[t][j] = price  [r][16*t + 8*g + j]   (t=0..3, j=0..7)
//   st[t][j] = strategy[r][16*t + 8*g + j]
// Per step we compute s_new^T = W_j (A, 4x4 frags from LDS) * helper1^T (B frags
// packed from pr/st registers), accumulate f32, bias+tanh, then update state,
// costs and wealth entirely in registers.
__global__ __launch_bounds__(256) void trade_wealth_kernel(
    const float* __restrict__ price_in,
    const float* __restrict__ strat_in,
    const float* __restrict__ wealth_in,
    const float* __restrict__ incr,   // [NSTEPS, K, 64]
    const float* __restrict__ W,      // [NSTEPS, 64, 128]
    const float* __restrict__ b,      // [NSTEPS, 64]
    float* __restrict__ out,          // [K]
    int K)
{
    // 16 (t,c) blocks x 32 lanes x 16 halves = 8192 halves = 16 KB
    __shared__ _Float16 wstage[16 * 32 * 16];

    const int tid  = threadIdx.x;
    const int wave = tid >> 5;
    const int lane = tid & 31;
    const int g    = lane >> 4;    // which 8-col group within each 16-col tile
    const int r    = lane & 15;    // batch row within the wave's 16-row tile
    const int r_glob = blockIdx.x * 128 + wave * 16 + r;

    float pr[4][8];
    float st[4][8];

    const float* prow = price_in + (size_t)r_glob * MDIM;
    const float* srow = strat_in + (size_t)r_glob * MDIM;
#pragma unroll
    for (int t = 0; t < 4; ++t) {
        const int cbase = 16 * t + 8 * g;
#pragma unroll
        for (int j = 0; j < 8; ++j) {
            pr[t][j] = prow[cbase + j];
            st[t][j] = srow[cbase + j];
        }
    }
    float wealth = wealth_in[r_glob];

    for (int step = 0; step < NSTEPS; ++step) {
        // ---- stage W_j into LDS in A-fragment layout (f32 -> f16) ----
        __syncthreads();  // previous step's LDS reads complete
        const float* Wj = W + (size_t)step * (MDIM * 2 * MDIM);
#pragma unroll
        for (int e2 = 0; e2 < 2; ++e2) {
            const int e   = tid * 2 + e2;   // 0..511 (16 blocks x 32 lanes)
            const int blk = e >> 5;
            const int ln  = e & 31;
            const int tt  = blk >> 2, cc = blk & 3;
            const int mm  = 16 * tt + (ln & 15);
            const int gg  = ln >> 4;
            const float* wrow = Wj + mm * 128 + 32 * cc + 8 * gg;
            v16h frag;
#pragma unroll
            for (int h = 0; h < 8; ++h)  frag[h] = (_Float16)wrow[h];       // K = base+0..7
#pragma unroll
            for (int h = 8; h < 16; ++h) frag[h] = (_Float16)wrow[h + 8];   // K = base+16..23
            *(v16h*)&wstage[(blk * 32 + ln) * 16] = frag;
        }
        __syncthreads();

        // ---- pack B fragments from current state: helper1 = [price | strategy] ----
        v16h bfrag[4];
#pragma unroll
        for (int c = 0; c < 2; ++c)
#pragma unroll
            for (int h = 0; h < 16; ++h)
                bfrag[c][h] = (_Float16)pr[2 * c + (h >= 8 ? 1 : 0)][h & 7];
#pragma unroll
        for (int c = 2; c < 4; ++c)
#pragma unroll
            for (int h = 0; h < 16; ++h)
                bfrag[c][h] = (_Float16)st[2 * (c - 2) + (h >= 8 ? 1 : 0)][h & 7];

        const float* irow = incr + ((size_t)step * K + r_glob) * MDIM;
        const float* brow = b + (size_t)step * MDIM;

        if (step + 1 < NSTEPS)  // hint the next stream tile (global_prefetch_b8)
            __builtin_prefetch(incr + ((size_t)(step + 1) * K + r_glob) * MDIM + 8 * g, 0, 1);

        float cost_p = 0.f, mult_p = 0.f;

#pragma unroll
        for (int t = 0; t < 4; ++t) {
            v8f acc = {};
#pragma unroll
            for (int c = 0; c < 4; ++c) {
                v16h afrag = *(const v16h*)&wstage[((t * 4 + c) * 32 + lane) * 16];
                acc = __builtin_amdgcn_wmma_f32_16x16x32_f16(
                    /*neg_a=*/false, afrag, /*neg_b=*/false, bfrag[c],
                    /*c_mod=*/(short)0, acc, /*reuse_a=*/false, /*reuse_b=*/false);
            }
            const int cbase = 16 * t + 8 * g;
            float inc[8];
#pragma unroll
            for (int j = 0; j < 8; ++j)
                inc[j] = __builtin_nontemporal_load(&irow[cbase + j]);  // streamed once
#pragma unroll
            for (int j = 0; j < 8; ++j) {
                float x = acc[j] + brow[cbase + j];
                float e = __expf(2.f * x);
                float s = 1.f - 2.f / (e + 1.f);          // tanh(x), saturates correctly
                float change = s - st[t][j];
                cost_p += fabsf(change) * pr[t][j];        // uses OLD price
                st[t][j] = s;
                mult_p  += s * inc[j];
                pr[t][j] += inc[j];
            }
        }

        // each row r lives in lanes r and r+16 -> one xor-16 reduction
        float cost = cost_p + __shfl_xor(cost_p, 16, 32);
        float mult = mult_p + __shfl_xor(mult_p, 16, 32);
        wealth += mult - 0.01f * cost;
    }

    // final costs: -0.01 * sum(|strategy| * price)
    float fc_p = 0.f;
#pragma unroll
    for (int t = 0; t < 4; ++t)
#pragma unroll
        for (int j = 0; j < 8; ++j)
            fc_p += fabsf(st[t][j]) * pr[t][j];
    float fc = fc_p + __shfl_xor(fc_p, 16, 32);

    if (g == 0)
        out[r_glob] = wealth - 0.01f * fc;
}

extern "C" void kernel_launch(void* const* d_in, const int* in_sizes, int n_in,
                              void* d_out, int out_size, void* d_ws, size_t ws_size,
                              hipStream_t stream) {
    const float* price  = (const float*)d_in[0];
    const float* strat  = (const float*)d_in[1];
    const float* wealth = (const float*)d_in[2];
    // d_in[3] = costs (unused by the reference computation)
    const float* incr   = (const float*)d_in[4];
    const float* W      = (const float*)d_in[5];
    const float* b      = (const float*)d_in[6];
    float* out = (float*)d_out;

    const int K = in_sizes[2];          // wealth has K elements
    const int grid = K / 128;           // 256 threads = 8 waves x 16 rows = 128 rows/block

    trade_wealth_kernel<<<grid, 256, 0, stream>>>(price, strat, wealth, incr, W, b, out, K);
}
